// GDEncoder_50431505989923
// MI455X (gfx1250) — matrix operand order repaired and checked by reading, not compiled, not run on hardware
//
#include <hip/hip_runtime.h>
#include <hip/hip_bf16.h>
#include <math.h>

// ---------------- problem constants ----------------
#define SEQ    640
#define BATCH  40
#define D_IN   8
#define D_TRAJ 32
#define D_LSTM 64
#define NH     4
#define DH     48
#define EMB    192           // NH*DH
#define MROWS  (SEQ*BATCH)   // 25600

typedef __attribute__((ext_vector_type(16))) __bf16 bf16x16;
typedef __attribute__((ext_vector_type(4)))  __bf16 bf16x4;
typedef __attribute__((ext_vector_type(8)))  float  f32x8;

// ---------------- WMMA helper (gfx1250 wave32, 16x16x32 bf16) ----------------
__device__ __forceinline__ f32x8 wmma_bf16(bf16x16 a, bf16x16 b, f32x8 c) {
  return __builtin_amdgcn_wmma_f32_16x16x32_bf16(
      /*neg_a=*/false, a, /*neg_b=*/false, b,
      /*c_mod=*/(short)0, c, /*reuse_a=*/false, /*reuse_b=*/false);
}

// ---------------- fragment layout helpers ----------------
// A (16x32 bf16) per-lane elements are two contiguous 8-element runs:
//   a[0..7]  = row[ 8*half ..  8*half+7]
//   a[8..15] = row[16+8*half .. 16+8*half+7]     (half = lane>>4, row = lane&15)
union AFrag { bf16x16 v; bf16x4 q[4]; };

// load A fragment from a row-major bf16 matrix (LDS or global), leading dim ld
__device__ __forceinline__ bf16x16 load_a_rm(const __bf16* base, int ld, int lane) {
  int half = lane >> 4, r = lane & 15;
  const __bf16* p = base + r * ld + half * 8;
  AFrag f;
  f.q[0] = *(const bf16x4*)(p);
  f.q[1] = *(const bf16x4*)(p + 4);
  f.q[2] = *(const bf16x4*)(p + 16);
  f.q[3] = *(const bf16x4*)(p + 20);
  return f.v;
}

// packed-fragment load: lane's 16 bf16 are contiguous (32B) -> 2x global_load_b128
__device__ __forceinline__ bf16x16 load_frag_packed(const __bf16* base, int block, int lane) {
  return *(const bf16x16*)(base + (((size_t)block * 32 + lane) << 4));
}

// flattened index of element (k,n) of a KxN B matrix in packed-fragment order
// (Nt = N/16).  B layout: lane 0-15 col n, K 0..7,16..23 ; lane 16-31 col n, K 8..15,24..31.
__device__ __forceinline__ int b_packed_idx(int k, int n, int Nt) {
  int kt = k >> 5, kr = k & 31, nt = n >> 4, nl = n & 15;
  int half = (kr >> 3) & 1, elem = (kr & 7) | ((kr >> 4) << 3);
  return (((kt * Nt + nt) * 32) + half * 16 + nl) * 16 + elem;
}

// flattened index of element (r,k) of an MxK A matrix in packed-fragment order
// (Kt = K/32).
__device__ __forceinline__ int a_packed_idx(int r, int k, int Kt) {
  int mt = r >> 4, rl = r & 15, kt = k >> 5, kloc = k & 31;
  int half = (kloc >> 3) & 1;
  int elem = (kloc & 7) | ((kloc >> 4) << 3);
  return (((mt * Kt + kt) * 32) + half * 16 + rl) * 16 + elem;
}

__device__ __forceinline__ float sigf(float x) { return 1.f / (1.f + expf(-x)); }

// ---------------- workspace layout (bytes) ----------------
static const size_t OFF_X    = 0;                                     // x f32 (25600x32)
static const size_t OFF_WCAT = OFF_X    + (size_t)MROWS * 32 * 4;     // bf16 96x256 packed
static const size_t OFF_WQKV = OFF_WCAT + (size_t)96 * 256 * 2;       // bf16 64x576 packed
static const size_t OFF_WAG  = OFF_WQKV + (size_t)64 * 576 * 2;       // bf16 192x128 packed
static const size_t OFF_SEQ  = OFF_WAG  + (size_t)192 * 128 * 2;      // seq f32 (25600x64)
static const size_t OFF_SEQB = OFF_SEQ  + (size_t)MROWS * 64 * 4;     // seq bf16 row-major
static const size_t OFF_QP   = OFF_SEQB + (size_t)MROWS * 64 * 2;     // q packed A-frags (2560x3072)
static const size_t OFF_KTP  = OFF_QP   + (size_t)2560 * 3072 * 2;    // kT packed B-frags
static const size_t OFF_VP   = OFF_KTP  + (size_t)2560 * 3072 * 2;    // v packed B-frags
static const size_t OFF_OBF  = OFF_VP   + (size_t)2560 * 3072 * 2;    // o bf16 (25600x192)
static const size_t WS_NEED  = OFF_OBF  + (size_t)MROWS * 192 * 2;

// ---------------- kernel 0: weight preconversion into packed-fragment bf16 ----------------
__global__ void prep_kernel(const float* W_ih, const float* W_hh,
                            const float* Wq, const float* Wk, const float* Wv,
                            const float* Wa, const float* Wg,
                            __bf16* Wcat, __bf16* Wqkv, __bf16* Wag) {
  int idx = blockIdx.x * blockDim.x + threadIdx.x;
  const int N1 = 96 * 256, N2 = 64 * 576, N3 = 192 * 128;
  if (idx < N1) {                           // Wcat: K=96, N=256, Nt=16
    int k = idx / 256, n = idx % 256;
    float v = (k < 32) ? W_ih[n * 32 + k] : W_hh[n * 64 + (k - 32)];
    Wcat[b_packed_idx(k, n, 16)] = (__bf16)v;
  } else if (idx < N1 + N2) {               // Wqkv: K=64, N=576, Nt=36
    int j = idx - N1;
    int k = j / 576, n = j % 576;
    float v = (n < 192) ? Wq[n * 64 + k]
            : (n < 384) ? Wk[(n - 192) * 64 + k]
                        : Wv[(n - 384) * 64 + k];
    Wqkv[b_packed_idx(k, n, 36)] = (__bf16)v;
  } else if (idx < N1 + N2 + N3) {          // Wag: K=192, N=128, Nt=8
    int j = idx - N1 - N2;
    int k = j / 128, n = j % 128;
    float v = (n < 64) ? Wa[n * 192 + k] : Wg[(n - 64) * 192 + k];
    Wag[b_packed_idx(k, n, 8)] = (__bf16)v;
  }
}

// ---------------- kernel 1: trajectory embedding + ELU ----------------
__global__ void embed_kernel(const float* hist, const float* W1, const float* b1,
                             float* x) {
  int idx = blockIdx.x * blockDim.x + threadIdx.x;
  if (idx >= MROWS * D_TRAJ) return;
  int row = idx >> 5, k = idx & 31;
  float s = b1[k];
#pragma unroll
  for (int f = 0; f < D_IN; ++f) s += hist[row * D_IN + f] * W1[k * D_IN + f];
  x[idx] = (s > 0.f) ? s : expm1f(s);
}

// ---------------- kernel 2: sequential LSTM, one workgroup, WMMA gates ----------------
// gates (40x256) = [x_t | h] (48x96 padded) @ Wcat (96x256 packed), 3 Ksteps.
// h lives in-place in the bf16 A-staging buffer columns 32..95.
__global__ __launch_bounds__(256) void lstm_kernel(
    const float* __restrict__ x, const __bf16* __restrict__ Wcat,
    const float* __restrict__ b_ih, const float* __restrict__ b_hh,
    float* __restrict__ seq, __bf16* __restrict__ seqbf) {
  __shared__ __bf16 As[48 * 96];    // [x_t | h], rows 40..47 stay zero
  __shared__ float  Gs[48 * 256];   // gate pre-activations (full 48 rows, unguarded)
  int tid = threadIdx.x, lane = tid & 31, wave = tid >> 5;

#if __has_builtin(__builtin_amdgcn_s_wait_tensorcnt)
  __builtin_amdgcn_s_wait_tensorcnt(0);
#endif

  float creg[10];                   // cell state: element i = tid + s*256
#pragma unroll
  for (int s = 0; s < 10; ++s) creg[s] = 0.f;
  for (int i = tid; i < 48 * 96; i += 256) As[i] = (__bf16)0.f;  // zero pad + h0
  __builtin_prefetch(Wcat, 0, 3);   // global_prefetch_b8 of the B matrix
  __syncthreads();

  for (int t = 0; t < SEQ; ++t) {
    // refresh x part of A (h part already in place from previous step)
    for (int i = tid; i < BATCH * 32; i += 256) {
      int r = i >> 5, k = i & 31;
      As[r * 96 + k] = (__bf16)x[(t * BATCH + r) * 32 + k];
    }
    __syncthreads();

    // 48 C-tiles (3 M x 16 N), 6 per wave, K = 96 -> 3 WMMA steps
    for (int tile = wave; tile < 48; tile += 8) {
      int mt = tile % 3, nt = tile / 3;
      f32x8 acc = {};
#pragma unroll
      for (int kk = 0; kk < 3; ++kk) {
        bf16x16 a = load_a_rm(&As[mt * 16 * 96 + kk * 32], 96, lane);
        bf16x16 b = load_frag_packed(Wcat, kk * 16 + nt, lane);
        acc = wmma_bf16(a, b, acc);
      }
      int half = lane >> 4, nl = lane & 15;
#pragma unroll
      for (int i = 0; i < 8; ++i)
        Gs[(mt * 16 + half * 8 + i) * 256 + nt * 16 + nl] = acc[i];
    }
    __syncthreads();

    // gate nonlinearity + state update (torch order i,f,g,o)
#pragma unroll
    for (int s = 0; s < 10; ++s) {
      int i = tid + s * 256;
      int r = i >> 6, j = i & 63;
      float gi = Gs[r * 256 + j]        + b_ih[j]        + b_hh[j];
      float gf = Gs[r * 256 + 64 + j]   + b_ih[64 + j]   + b_hh[64 + j];
      float gg = Gs[r * 256 + 128 + j]  + b_ih[128 + j]  + b_hh[128 + j];
      float go = Gs[r * 256 + 192 + j]  + b_ih[192 + j]  + b_hh[192 + j];
      float c = sigf(gf) * creg[s] + sigf(gi) * tanhf(gg);
      float h = sigf(go) * tanhf(c);
      creg[s] = c;
      __bf16 hb = (__bf16)h;
      As[r * 96 + 32 + j] = hb;                        // h in place for next step
      seq[(size_t)(t * BATCH + r) * 64 + j]   = h;
      seqbf[(size_t)(t * BATCH + r) * 64 + j] = hb;
    }
    __syncthreads();
  }
}

// ---------------- kernel 3: QKV projections -> packed fragment layouts ----------------
__global__ __launch_bounds__(256) void qkv_kernel(
    const __bf16* __restrict__ seqbf, const __bf16* __restrict__ Wqkv,
    const float* __restrict__ bq, const float* __restrict__ bk,
    const float* __restrict__ bv,
    __bf16* __restrict__ qP, __bf16* __restrict__ kTP, __bf16* __restrict__ vP) {
  int t = blockIdx.x;
  __shared__ __bf16 As[48 * 64];
  int tid = threadIdx.x, lane = tid & 31, wave = tid >> 5;
  for (int i = tid; i < 48 * 64; i += 256) {
    int r = i >> 6, k = i & 63;
    As[i] = (r < BATCH) ? seqbf[((size_t)t * BATCH + r) * 64 + k] : (__bf16)0.f;
  }
  __syncthreads();
  // 3 M-tiles x 36 N-tiles (N=576 = q|k|v), K=64 -> 2 WMMA steps
  for (int tile = wave; tile < 108; tile += 8) {
    int mt = tile % 3, nt = tile / 3;
    f32x8 acc = {};
#pragma unroll
    for (int kk = 0; kk < 2; ++kk) {
      bf16x16 a = load_a_rm(&As[mt * 16 * 64 + kk * 32], 64, lane);
      bf16x16 b = load_frag_packed(Wqkv, kk * 36 + nt, lane);
      acc = wmma_bf16(a, b, acc);
    }
    int half = lane >> 4, nl = lane & 15;
    int n = nt * 16 + nl;
    int region = n / 192, nn = n % 192, hh = nn / 48, d = nn % 48;
    float bias = (region == 0) ? bq[nn] : (region == 1) ? bk[nn] : bv[nn];
    size_t base = ((size_t)t * 4 + hh) * 3072;   // per-(t,head) packed tile
#pragma unroll
    for (int i = 0; i < 8; ++i) {
      int r = mt * 16 + half * 8 + i;
      if (r < BATCH) {
        __bf16 val = (__bf16)(acc[i] + bias);
        if (region == 0)      qP [base + a_packed_idx(r, d, 2)] = val; // A 48x64
        else if (region == 1) kTP[base + b_packed_idx(d, r, 3)] = val; // B 64x48
        else                  vP [base + b_packed_idx(r, d, 3)] = val; // B 64x48
      }
    }
  }
}

// ---------------- kernel 4: per-(t,head) attention, frags straight from global ----------------
__global__ __launch_bounds__(128) void attn_kernel(
    const __bf16* __restrict__ qP, const __bf16* __restrict__ kTP,
    const __bf16* __restrict__ vP, __bf16* __restrict__ obf) {
  int blk = blockIdx.x;          // t*NH + h
  int t = blk >> 2, h = blk & 3;
  __shared__ float  Sg[48 * 48];
  __shared__ __bf16 Pp[48 * 64];
  int tid = threadIdx.x, lane = tid & 31, wave = tid >> 5;
  const __bf16* qb = qP  + (size_t)blk * 3072;
  const __bf16* kb = kTP + (size_t)blk * 3072;
  const __bf16* vb = vP  + (size_t)blk * 3072;

  // scores S = Q(48x64) @ KT(64x48): 9 tiles, K=64 -> 2 WMMA steps
  for (int tile = wave; tile < 9; tile += 4) {
    int mt = tile / 3, nt = tile % 3;
    f32x8 acc = {};
#pragma unroll
    for (int kk = 0; kk < 2; ++kk) {
      bf16x16 a = load_frag_packed(qb, mt * 2 + kk, lane);
      bf16x16 b = load_frag_packed(kb, kk * 3 + nt, lane);
      acc = wmma_bf16(a, b, acc);
    }
    int half = lane >> 4, nl = lane & 15;
#pragma unroll
    for (int i = 0; i < 8; ++i)
      Sg[(mt * 16 + half * 8 + i) * 48 + nt * 16 + nl] = acc[i];
  }
  __syncthreads();

  // softmax over 40 valid cols, scale 1/8; build padded P (48x64, pads zero)
  if (tid < 48) {
    int r = tid;
    if (r < BATCH) {
      float mx = -1e30f;
      for (int c = 0; c < BATCH; ++c) mx = fmaxf(mx, Sg[r * 48 + c] * 0.125f);
      float e[BATCH], sum = 0.f;
      for (int c = 0; c < BATCH; ++c) { e[c] = expf(Sg[r * 48 + c] * 0.125f - mx); sum += e[c]; }
      float inv = 1.f / sum;
      for (int c = 0; c < 64; ++c)
        Pp[r * 64 + c] = (__bf16)((c < BATCH) ? e[c] * inv : 0.f);
    } else {
      for (int c = 0; c < 64; ++c) Pp[r * 64 + c] = (__bf16)0.f;
    }
  }
  __syncthreads();

  // O = P(48x64) @ V(64x48): 9 tiles, K=64 -> 2 WMMA steps
  for (int tile = wave; tile < 9; tile += 4) {
    int mt = tile / 3, nt = tile % 3;
    f32x8 acc = {};
#pragma unroll
    for (int kk = 0; kk < 2; ++kk) {
      bf16x16 a = load_a_rm(&Pp[mt * 16 * 64 + kk * 32], 64, lane);
      bf16x16 b = load_frag_packed(vb, kk * 3 + nt, lane);
      acc = wmma_bf16(a, b, acc);
    }
    int half = lane >> 4, nl = lane & 15;
#pragma unroll
    for (int i = 0; i < 8; ++i) {
      int r = mt * 16 + half * 8 + i, c = nt * 16 + nl;
      if (r < BATCH)
        obf[((size_t)t * BATCH + r) * 192 + h * 48 + c] = (__bf16)acc[i];
    }
  }
}

// ---------------- kernel 5: GLU + residual + LayerNorm ----------------
__global__ __launch_bounds__(256) void glu_ln_kernel(
    const __bf16* __restrict__ obf, const __bf16* __restrict__ Wag,
    const float* __restrict__ ba, const float* __restrict__ bg,
    const float* __restrict__ seqf, const float* __restrict__ gamma,
    const float* __restrict__ beta, float* __restrict__ out) {
  int row0 = blockIdx.x * 48;
  __shared__ __bf16 Ao[48 * 192];
  __shared__ float  Yc[48 * 128];
  int tid = threadIdx.x, lane = tid & 31, wave = tid >> 5;
  for (int i = tid; i < 48 * 192; i += 256) {
    int r = i / 192, k = i % 192;
    int row = row0 + r;
    Ao[i] = (row < MROWS) ? obf[(size_t)row * 192 + k] : (__bf16)0.f;
  }
  __syncthreads();
  // Y(48x128) = Ao(48x192) @ Wag(192x128 packed): 24 tiles, K=192 -> 6 steps
  for (int tile = wave; tile < 24; tile += 8) {
    int mt = tile % 3, nt = tile / 3;
    f32x8 acc = {};
#pragma unroll
    for (int kk = 0; kk < 6; ++kk) {
      bf16x16 a = load_a_rm(&Ao[mt * 16 * 192 + kk * 32], 192, lane);
      bf16x16 b = load_frag_packed(Wag, kk * 8 + nt, lane);
      acc = wmma_bf16(a, b, acc);
    }
    int half = lane >> 4, nl = lane & 15;
#pragma unroll
    for (int i = 0; i < 8; ++i)
      Yc[(mt * 16 + half * 8 + i) * 128 + nt * 16 + nl] = acc[i];
  }
  __syncthreads();
  if (tid < 48) {
    int r = tid, row = row0 + r;
    if (row < MROWS) {
      float y[64], mu = 0.f;
#pragma unroll
      for (int d = 0; d < 64; ++d) {
        float a = Yc[r * 128 + d] + ba[d];
        float g = Yc[r * 128 + 64 + d] + bg[d];
        y[d] = seqf[(size_t)row * 64 + d] + a * sigf(g);
        mu += y[d];
      }
      mu *= (1.f / 64.f);
      float var = 0.f;
#pragma unroll
      for (int d = 0; d < 64; ++d) { float z = y[d] - mu; var += z * z; }
      var *= (1.f / 64.f);
      float inv = rsqrtf(var + 1e-5f);
#pragma unroll
      for (int d = 0; d < 64; ++d)
        out[(size_t)row * 64 + d] = (y[d] - mu) * inv * gamma[d] + beta[d];
    }
  }
}

// ---------------- launcher ----------------
extern "C" void kernel_launch(void* const* d_in, const int* in_sizes, int n_in,
                              void* d_out, int out_size, void* d_ws, size_t ws_size,
                              hipStream_t stream) {
  const float* hist  = (const float*)d_in[0];
  // d_in[1] = adj (unused, use_spatial=False)
  const float* W1    = (const float*)d_in[2];
  const float* b1    = (const float*)d_in[3];
  const float* W_ih  = (const float*)d_in[4];
  const float* W_hh  = (const float*)d_in[5];
  const float* b_ih  = (const float*)d_in[6];
  const float* b_hh  = (const float*)d_in[7];
  const float* Wq    = (const float*)d_in[8];
  const float* bq    = (const float*)d_in[9];
  const float* Wk    = (const float*)d_in[10];
  const float* bk    = (const float*)d_in[11];
  const float* Wv    = (const float*)d_in[12];
  const float* bv    = (const float*)d_in[13];
  const float* Wa    = (const float*)d_in[14];
  const float* ba    = (const float*)d_in[15];
  const float* Wg    = (const float*)d_in[16];
  const float* bg    = (const float*)d_in[17];
  const float* gamma = (const float*)d_in[18];
  const float* beta  = (const float*)d_in[19];

  if (ws_size < WS_NEED) return;  // insufficient scratch; nothing safe to do

  char* ws = (char*)d_ws;
  float*  x     = (float*)(ws + OFF_X);
  __bf16* Wcat  = (__bf16*)(ws + OFF_WCAT);
  __bf16* Wqkv  = (__bf16*)(ws + OFF_WQKV);
  __bf16* Wag   = (__bf16*)(ws + OFF_WAG);
  float*  seqf  = (float*)(ws + OFF_SEQ);
  __bf16* seqbf = (__bf16*)(ws + OFF_SEQB);
  __bf16* qP    = (__bf16*)(ws + OFF_QP);
  __bf16* kTP   = (__bf16*)(ws + OFF_KTP);
  __bf16* vP    = (__bf16*)(ws + OFF_VP);
  __bf16* obf   = (__bf16*)(ws + OFF_OBF);
  float*  out   = (float*)d_out;

  // zero the packed q/kT/v region so all WMMA K-padding is exactly 0
  hipMemsetAsync(qP, 0, OFF_OBF - OFF_QP, stream);

  // weight preconversion into packed-fragment bf16 layouts
  {
    int total = 96 * 256 + 64 * 576 + 192 * 128;
    prep_kernel<<<(total + 255) / 256, 256, 0, stream>>>(
        W_ih, W_hh, Wq, Wk, Wv, Wa, Wg, Wcat, Wqkv, Wag);
  }
  // trajectory embedding + ELU
  embed_kernel<<<(MROWS * D_TRAJ + 255) / 256, 256, 0, stream>>>(hist, W1, b1, x);
  // sequential LSTM (single persistent workgroup)
  lstm_kernel<<<1, 256, 0, stream>>>(x, Wcat, b_ih, b_hh, seqf, seqbf);
  // QKV projections -> packed fragment tiles
  qkv_kernel<<<SEQ, 256, 0, stream>>>(seqbf, Wqkv, bq, bk, bv, qP, kTP, vP);
  // attention per (t, head)
  attn_kernel<<<SEQ * NH, 128, 0, stream>>>(qP, kTP, vP, obf);
  // GLU + residual + LayerNorm
  glu_ln_kernel<<<(MROWS + 47) / 48, 256, 0, stream>>>(
      obf, Wag, ba, bg, seqf, gamma, beta, out);
}
